// Attention_9199819948554
// MI455X (gfx1250) — matrix operand order repaired and checked by reading, not compiled, not run on hardware
//
#include <hip/hip_runtime.h>
#include <cstdint>

// ---------------------------------------------------------------------------
// CDNA5 (gfx1250) wave32 WMMA attention.
//   v_wmma_f32_16x16x32_bf16 :  D(16x16,f32) = A(16x32,bf16) x B(32x16,bf16) + C
// A layout (16-bit A 16x32): lane(0-15)=row M; VGPR i holds
//   K = half*16 + 2i,2i+1 (same k order as B)
// B layout (32x16): lane = col N; VGPR i holds K = half*16 + 2i, 2i+1
// C/D layout: lane = col N; VGPR i holds row M = i + half*8
// ---------------------------------------------------------------------------

typedef __attribute__((ext_vector_type(16))) __bf16 v16bf;
typedef __attribute__((ext_vector_type(8)))  float  v8f;

union V16U { v16bf v; uint32_t u[8]; };

__device__ __forceinline__ uint16_t f2bf(float f) {
  union { float f; uint32_t u; } c; c.f = f;
  uint32_t r = c.u + 0x7fffu + ((c.u >> 16) & 1u);   // round-to-nearest-even
  return (uint16_t)(r >> 16);
}

// Load a 16x32 bf16 A-tile (row-major, leading dim ld elems) into WMMA A layout.
__device__ __forceinline__ v16bf load_A16x32(const uint16_t* __restrict__ base, int ld) {
  const int lane = threadIdx.x & 31;
  const int half = lane >> 4;
  const int row  = lane & 15;
  const uint16_t* r = base + row * ld + half * 16;
  V16U t;
#pragma unroll
  for (int i = 0; i < 8; ++i) {
    t.u[i]     = *(const uint32_t*)(r + 2 * i);        // K = half*16 + 2i,2i+1
  }
  return t.v;
}

// Load a 32x16 bf16 B-tile where B[k][n] lives at base[n*ld + k] (k contiguous).
__device__ __forceinline__ v16bf load_B32x16(const uint16_t* __restrict__ base, int ld) {
  const int lane = threadIdx.x & 31;
  const int half = lane >> 4;
  const int col  = lane & 15;
  const uint16_t* c = base + col * ld + half * 16;
  V16U t;
#pragma unroll
  for (int i = 0; i < 8; ++i) t.u[i] = *(const uint32_t*)(c + 2 * i);  // K = half*16 + 2i
  return t.v;
}

__device__ __forceinline__ v8f wmma_bf16(v16bf a, v16bf b, v8f c) {
  return __builtin_amdgcn_wmma_f32_16x16x32_bf16(false, a, false, b, (short)0, c, false, false);
}

// --- CDNA5 async global->LDS (tracked by ASYNCcnt, see CDNA5 ISA) --------
// GVS mode: mem addr = SGPR64 base + signed 32-bit VGPR byte offset.
// VDST VGPR carries the LDS byte address (low 32 bits of the flat aperture).
__device__ __forceinline__ void async_ld_b128(uint32_t lds_byte, uint32_t g_byte,
                                              const void* sbase) {
  asm volatile("global_load_async_to_lds_b128 %0, %1, %2"
               :: "v"(lds_byte), "v"(g_byte), "s"(sbase) : "memory");
}
__device__ __forceinline__ void wait_async0() {
  asm volatile("s_wait_asynccnt 0x0" ::: "memory");
}
__device__ __forceinline__ uint32_t lds_addr_of(const void* p) {
  return (uint32_t)(uintptr_t)p;   // flat LDS aperture: low 32 bits are the LDS address
}

// ------------------------- conversion / transpose kernels -------------------

__global__ void cvt_bf16_kernel(const float* __restrict__ in, uint16_t* __restrict__ out, int n) {
  int i = blockIdx.x * blockDim.x + threadIdx.x;
  if (i < n) out[i] = f2bf(in[i]);
}

// W: (H=16, D=1024, K=64) fp32  ->  wt[(h*64+k)*1024 + d] bf16
__global__ void transpose_w_kernel(const float* __restrict__ w, uint16_t* __restrict__ wt) {
  int i = blockIdx.x * blockDim.x + threadIdx.x;      // 1,048,576 elems
  int k = i & 63, d = (i >> 6) & 1023, h = i >> 16;
  wt[(size_t)(h * 64 + k) * 1024 + d] = f2bf(w[i]);
}

// W_O: (HK=1024, D=1024) fp32 -> wt[d*1024 + hk] bf16
__global__ void transpose_wo_kernel(const float* __restrict__ w, uint16_t* __restrict__ wt) {
  int i = blockIdx.x * blockDim.x + threadIdx.x;      // 1,048,576 elems
  int d = i & 1023, hk = i >> 10;
  wt[(size_t)d * 1024 + hk] = f2bf(w[i]);
}

// ------------------------------ QKV projection ------------------------------
// grid = (B*S/128 = 32, H = 16, 3), block = 128 (4 waves).
// Block computes a 128x64 tile; the 32x64 bf16 B tile per K-step is staged in
// double-buffered LDS with global_load_async_to_lds_b128, shared by 4 waves.
// Q,K stored (B,H,S,64); V stored transposed (B,H,64,S) for the PV B-operand.
__global__ __launch_bounds__(128) void qkv_proj_kernel(
    const uint16_t* __restrict__ xb,
    const uint16_t* __restrict__ wqt, const uint16_t* __restrict__ wkt,
    const uint16_t* __restrict__ wvt,
    const float* __restrict__ bQ, const float* __restrict__ bK, const float* __restrict__ bV,
    uint16_t* __restrict__ Qb, uint16_t* __restrict__ Kb, uint16_t* __restrict__ Vt)
{
  __shared__ __align__(16) uint16_t ldsB[2][64 * 32];   // [N=64][K=32], double buf
  const int z = blockIdx.z;
  const uint16_t* wt  = (z == 0) ? wqt : (z == 1) ? wkt : wvt;
  const float*   bias = (z == 0) ? bQ  : (z == 1) ? bK  : bV;
  uint16_t*      outp = (z == 0) ? Qb  : (z == 1) ? Kb  : Vt;
  const int h   = blockIdx.y;
  const int bs  = blockIdx.x;
  const int b   = bs >> 4;                       // 16 blocks of 128 rows per batch
  const int tid = threadIdx.x;
  const int wave = tid >> 5;
  const int s0  = (bs & 15) * 128 + wave * 32;   // this wave's 32 rows
  const uint32_t lds0 = lds_addr_of(&ldsB[0][0]);

  // Async-fill one 64x(32 bf16) B tile: 4 KB = 256 x 16B, 2 transfers/thread.
  auto fill = [&](int buf, int k0) {
#pragma unroll
    for (int j = 0; j < 2; ++j) {
      const int id  = tid + 128 * j;             // 0..255
      const int row = id >> 2;                   // N index 0..63
      const int ch  = id & 3;                    // 16B chunk within 64B row
      const uint32_t lb = lds0 + (uint32_t)(buf * 4096 + row * 64 + ch * 16);
      const uint32_t gb = (uint32_t)(((h * 64 + row) * 1024 + k0) * 2 + ch * 16);
      async_ld_b128(lb, gb, wt);
    }
  };

  fill(0, 0);                                    // prologue
  v8f acc[2][4] = {};
  const uint16_t* abase = xb + (size_t)(b * 2048 + s0) * 1024;
  for (int k0 = 0; k0 < 1024; k0 += 32) {
    const int buf = (k0 >> 5) & 1;
    wait_async0();                               // my slice of buf is in LDS
    __syncthreads();                             // everyone's slice; prev reads done
    if (k0 + 32 < 1024) fill(buf ^ 1, k0 + 32);  // overlap next fill with compute
    v16bf a0 = load_A16x32(abase + k0, 1024);
    v16bf a1 = load_A16x32(abase + 16 * 1024 + k0, 1024);
    v16bf bm[4];
#pragma unroll
    for (int t = 0; t < 4; ++t)                  // all 8 ds_load_b128 up front
      bm[t] = load_B32x16(&ldsB[buf][t * 16 * 32], 32);
#pragma unroll
    for (int t = 0; t < 4; ++t) {                // 8 back-to-back WMMAs
      acc[0][t] = wmma_bf16(a0, bm[t], acc[0][t]);
      acc[1][t] = wmma_bf16(a1, bm[t], acc[1][t]);
    }
  }
  const int lane = tid & 31, half = lane >> 4, n = lane & 15;
  const int bh = b * 16 + h;
#pragma unroll
  for (int sub = 0; sub < 2; ++sub) {
#pragma unroll
    for (int t = 0; t < 4; ++t) {
#pragma unroll
      for (int i = 0; i < 8; ++i) {
        const int row = s0 + sub * 16 + i + half * 8;
        const int col = t * 16 + n;
        float v = acc[sub][t][i] + bias[h * 64 + col];
        if (z < 2) outp[(size_t)(bh * 2048 + row) * 64 + col]   = f2bf(v);
        else       outp[(size_t)(bh * 64 + col) * 2048 + row]   = f2bf(v);
      }
    }
  }
}

// ------------------------------ flash attention -----------------------------
// grid = (B*S/32 = 128, H = 16), block = 32. One wave owns 32 queries (two
// 16-row sub-tiles) so each K/V tile is reused twice. Causal online softmax;
// P goes C-layout -> LDS(bf16) -> A-layout for the PV WMMAs.
__global__ __launch_bounds__(32) void flash_attn_kernel(
    const uint16_t* __restrict__ Qb, const uint16_t* __restrict__ Kb,
    const uint16_t* __restrict__ Vt, uint16_t* __restrict__ Zb)
{
  __shared__ __align__(16) uint16_t pshm[32 * 32];
  const int h  = blockIdx.y;
  const int bs = blockIdx.x;
  const int b  = bs >> 6;                    // 64 query-tiles of 32 per batch
  const int q0 = (bs & 63) << 5;
  const int bh = b * 16 + h;
  const int lane = threadIdx.x & 31, half = lane >> 4, n = lane & 15;

  const uint16_t* qbase = Qb + (size_t)(bh * 2048 + q0) * 64;
  v16bf qa[2][2];
#pragma unroll
  for (int r = 0; r < 2; ++r) {
    qa[r][0] = load_A16x32(qbase + r * 16 * 64, 64);        // head-dim 0..31
    qa[r][1] = load_A16x32(qbase + r * 16 * 64 + 32, 64);   // head-dim 32..63
  }

  v8f acc[2][4] = {};
  float mrow[2][8], lrow[2][8];
#pragma unroll
  for (int r = 0; r < 2; ++r)
#pragma unroll
    for (int i = 0; i < 8; ++i) { mrow[r][i] = -__builtin_inff(); lrow[r][i] = 0.f; }

  const float scale = 0.125f;                // 1/sqrt(64)
  const int kend = q0 + 32;                  // exclusive causal key limit
  for (int kb = 0; kb < kend; kb += 32) {
    const uint16_t* kbase = Kb + (size_t)(bh * 2048 + kb) * 64;
    const v16bf k00 = load_B32x16(kbase, 64);                 // keys kb..+15, hd 0..31
    const v16bf k01 = load_B32x16(kbase + 32, 64);            // hd 32..63
    const v16bf k10 = load_B32x16(kbase + 16 * 64, 64);       // keys kb+16..+31
    const v16bf k11 = load_B32x16(kbase + 16 * 64 + 32, 64);
#pragma unroll
    for (int r = 0; r < 2; ++r) {
      v8f s0 = {}, s1 = {};
      s0 = wmma_bf16(qa[r][0], k00, s0);
      s0 = wmma_bf16(qa[r][1], k01, s0);
      s1 = wmma_bf16(qa[r][0], k10, s1);
      s1 = wmma_bf16(qa[r][1], k11, s1);
      // mask + online softmax (row stats via 16-lane xor-shuffle tree)
#pragma unroll
      for (int i = 0; i < 8; ++i) {
        const int qi = q0 + r * 16 + i + half * 8;
        float x0 = s0[i] * scale; if (kb + n      > qi) x0 = -__builtin_inff();
        float x1 = s1[i] * scale; if (kb + 16 + n > qi) x1 = -__builtin_inff();
        float rmax = fmaxf(x0, x1);
        rmax = fmaxf(rmax, __shfl_xor(rmax, 1, 32));
        rmax = fmaxf(rmax, __shfl_xor(rmax, 2, 32));
        rmax = fmaxf(rmax, __shfl_xor(rmax, 4, 32));
        rmax = fmaxf(rmax, __shfl_xor(rmax, 8, 32));
        const float mnew  = fmaxf(mrow[r][i], rmax);
        const float alpha = __expf(mrow[r][i] - mnew);
        const float p0 = __expf(x0 - mnew);
        const float p1 = __expf(x1 - mnew);
        float ps = p0 + p1;
        ps += __shfl_xor(ps, 1, 32);
        ps += __shfl_xor(ps, 2, 32);
        ps += __shfl_xor(ps, 4, 32);
        ps += __shfl_xor(ps, 8, 32);
        lrow[r][i] = lrow[r][i] * alpha + ps;
        mrow[r][i] = mnew;
#pragma unroll
        for (int t = 0; t < 4; ++t) acc[r][t][i] *= alpha;
        const int row = r * 16 + i + half * 8;
        pshm[row * 32 + n]      = f2bf(p0);
        pshm[row * 32 + 16 + n] = f2bf(p1);
      }
    }
    __syncthreads();
    const v16bf pa0 = load_A16x32(&pshm[0], 32);
    const v16bf pa1 = load_A16x32(&pshm[16 * 32], 32);
    const uint16_t* vbase = Vt + (size_t)(bh * 64) * 2048 + kb;
    v16bf vb[4];
#pragma unroll
    for (int t = 0; t < 4; ++t)                  // hoist all V fragment loads
      vb[t] = load_B32x16(vbase + (size_t)(t * 16) * 2048, 2048);
#pragma unroll
    for (int t = 0; t < 4; ++t) {
      acc[0][t] = wmma_bf16(pa0, vb[t], acc[0][t]);
      acc[1][t] = wmma_bf16(pa1, vb[t], acc[1][t]);
    }
    __syncthreads();
  }
  // normalize and emit Z in (B, S, H*64) bf16 for the output GEMM
#pragma unroll
  for (int r = 0; r < 2; ++r) {
    float inv[8];
#pragma unroll
    for (int i = 0; i < 8; ++i) inv[i] = 1.f / lrow[r][i];
#pragma unroll
    for (int t = 0; t < 4; ++t) {
#pragma unroll
      for (int i = 0; i < 8; ++i) {
        const int row = q0 + r * 16 + i + half * 8;
        Zb[(size_t)(b * 2048 + row) * 1024 + h * 64 + t * 16 + n] =
            f2bf(acc[r][t][i] * inv[i]);
      }
    }
  }
}

// ------------------------------ output projection ---------------------------
// grid = (B*S/128 = 32, D/64 = 16), block = 128 (4 waves). Same async-LDS
// B-staging as qkv_proj. out = Z @ Wo + bO (fp32 out).
__global__ __launch_bounds__(128) void out_proj_kernel(
    const uint16_t* __restrict__ Zb, const uint16_t* __restrict__ wot,
    const float* __restrict__ bO, float* __restrict__ out)
{
  __shared__ __align__(16) uint16_t ldsB[2][64 * 32];
  const int ng  = blockIdx.y;
  const int tid = threadIdx.x;
  const int wave = tid >> 5;
  const int r0  = blockIdx.x * 128 + wave * 32;
  const uint32_t lds0 = lds_addr_of(&ldsB[0][0]);

  auto fill = [&](int buf, int k0) {
#pragma unroll
    for (int j = 0; j < 2; ++j) {
      const int id  = tid + 128 * j;
      const int row = id >> 2;
      const int ch  = id & 3;
      const uint32_t lb = lds0 + (uint32_t)(buf * 4096 + row * 64 + ch * 16);
      const uint32_t gb = (uint32_t)(((ng * 64 + row) * 1024 + k0) * 2 + ch * 16);
      async_ld_b128(lb, gb, wot);
    }
  };

  fill(0, 0);
  v8f acc[2][4] = {};
  const uint16_t* abase = Zb + (size_t)r0 * 1024;
  for (int k0 = 0; k0 < 1024; k0 += 32) {
    const int buf = (k0 >> 5) & 1;
    wait_async0();
    __syncthreads();
    if (k0 + 32 < 1024) fill(buf ^ 1, k0 + 32);
    v16bf a0 = load_A16x32(abase + k0, 1024);
    v16bf a1 = load_A16x32(abase + 16 * 1024 + k0, 1024);
    v16bf bm[4];
#pragma unroll
    for (int t = 0; t < 4; ++t)
      bm[t] = load_B32x16(&ldsB[buf][t * 16 * 32], 32);
#pragma unroll
    for (int t = 0; t < 4; ++t) {
      acc[0][t] = wmma_bf16(a0, bm[t], acc[0][t]);
      acc[1][t] = wmma_bf16(a1, bm[t], acc[1][t]);
    }
  }
  const int lane = tid & 31, half = lane >> 4, n = lane & 15;
#pragma unroll
  for (int sub = 0; sub < 2; ++sub) {
#pragma unroll
    for (int t = 0; t < 4; ++t) {
#pragma unroll
      for (int i = 0; i < 8; ++i) {
        const int col = ng * 64 + t * 16 + n;
        out[(size_t)(r0 + sub * 16 + i + half * 8) * 1024 + col] =
            acc[sub][t][i] + bO[col];
      }
    }
  }
}

// ---------------------------------------------------------------------------

extern "C" void kernel_launch(void* const* d_in, const int* in_sizes, int n_in,
                              void* d_out, int out_size, void* d_ws, size_t ws_size,
                              hipStream_t stream) {
  const float* x   = (const float*)d_in[0];
  const float* W_Q = (const float*)d_in[1];
  const float* W_K = (const float*)d_in[2];
  const float* W_V = (const float*)d_in[3];
  const float* W_O = (const float*)d_in[4];
  const float* b_Q = (const float*)d_in[5];
  const float* b_K = (const float*)d_in[6];
  const float* b_V = (const float*)d_in[7];
  const float* b_O = (const float*)d_in[8];
  float* out = (float*)d_out;

  // Workspace layout (bf16 elements): 48 MiB total.
  uint16_t* ws = (uint16_t*)d_ws;
  const size_t NX = (size_t)2 * 2048 * 1024;   // 4,194,304 (x / Q / K / V / Z)
  const size_t NW = (size_t)1024 * 1024;       // 1,048,576 (each weight)
  uint16_t* xb  = ws;
  uint16_t* wqt = xb  + NX;
  uint16_t* wkt = wqt + NW;
  uint16_t* wvt = wkt + NW;
  uint16_t* wot = wvt + NW;
  uint16_t* Qb  = wot + NW;
  uint16_t* Kb  = Qb  + NX;
  uint16_t* Vt  = Kb  + NX;
  uint16_t* Zb  = Vt  + NX;

  cvt_bf16_kernel<<<(int)(NX / 256), 256, 0, stream>>>(x, xb, (int)NX);
  transpose_w_kernel<<<(int)(NW / 256), 256, 0, stream>>>(W_Q, wqt);
  transpose_w_kernel<<<(int)(NW / 256), 256, 0, stream>>>(W_K, wkt);
  transpose_w_kernel<<<(int)(NW / 256), 256, 0, stream>>>(W_V, wvt);
  transpose_wo_kernel<<<(int)(NW / 256), 256, 0, stream>>>(W_O, wot);

  dim3 gq(32, 16, 3);
  qkv_proj_kernel<<<gq, 128, 0, stream>>>(xb, wqt, wkt, wvt, b_Q, b_K, b_V, Qb, Kb, Vt);

  dim3 gf(128, 16);
  flash_attn_kernel<<<gf, 32, 0, stream>>>(Qb, Kb, Vt, Zb);

  dim3 go(32, 16);
  out_proj_kernel<<<go, 128, 0, stream>>>(Zb, wot, b_O, out);
}
